// PixelMultiheadAttention_34136400068728
// MI455X (gfx1250) — matrix operand order repaired and checked by reading, not compiled, not run on hardware
//
#include <hip/hip_runtime.h>
#include <hip/hip_bf16.h>

#define DEV static __device__ __forceinline__

typedef __attribute__((ext_vector_type(16))) __bf16       v16bf;
typedef __attribute__((ext_vector_type(8)))  float        v8f;
typedef __attribute__((ext_vector_type(4)))  unsigned int v4u;
typedef __attribute__((ext_vector_type(8)))  int          v8i;
typedef __attribute__((ext_vector_type(4)))  int          v4i;

// ---------------- constants -------------------------------------------------
// N=262144, LQ=1, LK=3, D=128, H=8, HD=16
#define NPIX   262144
#define DMODEL 128
#define LKTOK  3
#define PIX_PER_WAVE 16
#define WAVES_PER_BLK 4
#define PIX_PER_BLK  (PIX_PER_WAVE * WAVES_PER_BLK)   // 64

// ---------------- kernel 1: weights fp32 -> bf16 in d_ws --------------------
// d_ws layout (bf16, row-major): [Wq 128x128 | Wk | Wv | Wo] = 65536 elems
__global__ void cvt_weights_kernel(const float* __restrict__ in_w,   // 384x128
                                   const float* __restrict__ out_w,  // 128x128
                                   __bf16* __restrict__ ws) {
  int i = blockIdx.x * blockDim.x + threadIdx.x;   // pair index, 32768 total
  float2 p;
  if (i < 24576) p = ((const float2*)in_w)[i];
  else           p = ((const float2*)out_w)[i - 24576];
  ws[2 * i]     = (__bf16)p.x;     // native v_cvt_pk_bf16_f32 path
  ws[2 * i + 1] = (__bf16)p.y;
}

// ---------------- WMMA operand loaders --------------------------------------
// A (16x32 bf16) per ISA: lanes 0-15 hold K=0..7 (v0-3) and K=16..23 (v4-7);
// lanes 16-31 hold K=8..15 and K=24..31.  Source row is fp32 in global.
DEV v16bf make_a_from_f32(const float* __restrict__ row, int kb, int lane) {
  const int b0 = kb * 32 + ((lane >> 4) * 8);
  float4 x0 = *(const float4*)(row + b0);
  float4 x1 = *(const float4*)(row + b0 + 4);
  float4 y0 = *(const float4*)(row + b0 + 16);
  float4 y1 = *(const float4*)(row + b0 + 20);
  v16bf r;
  r[0]  = (__bf16)x0.x; r[1]  = (__bf16)x0.y; r[2]  = (__bf16)x0.z; r[3]  = (__bf16)x0.w;
  r[4]  = (__bf16)x1.x; r[5]  = (__bf16)x1.y; r[6]  = (__bf16)x1.z; r[7]  = (__bf16)x1.w;
  r[8]  = (__bf16)y0.x; r[9]  = (__bf16)y0.y; r[10] = (__bf16)y0.z; r[11] = (__bf16)y0.w;
  r[12] = (__bf16)y1.x; r[13] = (__bf16)y1.y; r[14] = (__bf16)y1.z; r[15] = (__bf16)y1.w;
  return r;
}

// A operand from bf16 row-major LDS (for the output projection)
DEV v16bf make_a_from_bf16(const __bf16* row, int kb, int lane) {
  const int b0 = kb * 32 + ((lane >> 4) * 8);
  union { v16bf v; uint4 q[2]; } r;
  r.q[0] = *(const uint4*)(row + b0);        // K b0..b0+7   -> ds_load_b128
  r.q[1] = *(const uint4*)(row + b0 + 16);   // K b0+16..+23 -> ds_load_b128
  return r.v;
}

// B (32x16 bf16): lanes 0-15 hold K=0..7 (v0-3) and K=16..23 (v4-7); lanes 16-31 hold
// K=8..15 and K=24..31, col = lane&15.  B[k][n] = W[n][k] with W row-major in LDS.
DEV v16bf make_b_from_lds(const __bf16* w, int j, int kb, int lane) {
  const int row = j * 16 + (lane & 15);             // output feature
  const int k0  = kb * 32 + ((lane >> 4) * 8);      // k0..k0+7 and k0+16..k0+23
  union { v16bf v; uint4 q[2]; } r;
  r.q[0] = *(const uint4*)(w + row * 128 + k0);     // ds_load_b128
  r.q[1] = *(const uint4*)(w + row * 128 + k0 + 16); // ds_load_b128
  return r.v;
}

// C/D tile (16x16 f32): VGPR v, lane ln -> row v+8*(ln>>4), col ln&15.
// Store as bf16 row-major into LDS staging.
DEV void store_tile_rowmajor(__bf16* dst, int j, int lane, v8f c, int row_base) {
  const int col = j * 16 + (lane & 15);
  const int r0  = row_base + ((lane >> 4) * 8);
#pragma unroll
  for (int v = 0; v < 8; ++v) dst[(r0 + v) * 128 + col] = (__bf16)c[v];
}

// One 16x128 projection: dst rows [row_base, row_base+16)
DEV void gemm_proj(const float* __restrict__ arow, const __bf16* w,
                   const float* __restrict__ bias, __bf16* dst,
                   int row_base, int lane) {
  v16bf a[4];
#pragma unroll
  for (int kb = 0; kb < 4; ++kb) a[kb] = make_a_from_f32(arow, kb, lane);
  for (int j = 0; j < 8; ++j) {
    const float bj = bias[j * 16 + (lane & 15)];
    v8f c = {bj, bj, bj, bj, bj, bj, bj, bj};
#pragma unroll
    for (int kb = 0; kb < 4; ++kb) {
      v16bf b = make_b_from_lds(w, j, kb, lane);
      c = __builtin_amdgcn_wmma_f32_16x16x32_bf16(false, a[kb], false, b,
                                                  (short)0, c, false, false);
    }
    store_tile_rowmajor(dst, j, lane, c, row_base);
  }
}

DEV void load16(const __bf16* src, float* dst) {
  union { uint4 q[2]; __bf16 b[16]; } r;
  r.q[0] = *(const uint4*)src;            // ds_load_b128
  r.q[1] = *(const uint4*)(src + 8);      // ds_load_b128
#pragma unroll
  for (int i = 0; i < 16; ++i) dst[i] = (float)r.b[i];
}

// ---------------- TDM: DMA one 32KB weight matrix from L2 into LDS ----------
DEV void tdm_load_32k(const __bf16* gsrc, __bf16* lds_dst) {
  __syncthreads();                    // prior readers of lds_dst are done
  if (threadIdx.x < 32) {             // one wave issues the tensor DMA
    const unsigned lds = (unsigned)(unsigned long long)lds_dst;   // LDS byte offset
    const unsigned long long ga = (unsigned long long)gsrc;
    v4u g0 = { 1u,                                     // count=1, user descriptor
               lds,                                    // lds_addr
               (unsigned)(ga & 0xFFFFFFFFu),           // global_addr[31:0]
               (unsigned)((ga >> 32) & 0x01FFFFFFull) | (2u << 30) }; // [56:32] | type=2
    // 1-D copy: data_size=8B, tensor_dim0=tile_dim0=4096 units (32KB)
    v8i g1 = { (int)(3u << 16),        // data_size=3 (8 bytes)
               (int)(4096u << 16),     // tensor_dim0[15:0]
               (int)(1u << 16),        // tensor_dim1 = 1
               (int)(4096u << 16),     // tile_dim0 = 4096
               1,                      // tile_dim1 = 1
               4096,                   // tensor_dim0_stride
               0, 0 };
    v4i gz4 = {0, 0, 0, 0};
    v8i gz8 = {0, 0, 0, 0, 0, 0, 0, 0};
    __builtin_amdgcn_tensor_load_to_lds(g0, g1, gz4, gz4, gz8, 0);
    __builtin_amdgcn_s_wait_tensorcnt(0);
  }
  __syncthreads();                    // weights visible to all waves
}

// ---------------- kernel 2: fused MHA ----------------------------------------
__global__ __launch_bounds__(128, 1)
void fused_mha_kernel(const float* __restrict__ Q, const float* __restrict__ K,
                      const float* __restrict__ V, const float* __restrict__ in_b,
                      const float* __restrict__ out_b,
                      const __bf16* __restrict__ wbf,
                      float* __restrict__ out) {
  __shared__ __bf16 sW[128 * 128];                       // 32 KB (current weight)
  __shared__ __bf16 sQ[WAVES_PER_BLK][16 * 128];         // 4x 4 KB
  __shared__ __bf16 sK[WAVES_PER_BLK][48 * 128];         // 4x 12 KB
  __shared__ __bf16 sV[WAVES_PER_BLK][48 * 128];         // 4x 12 KB
  __shared__ __bf16 sO[WAVES_PER_BLK][16 * 128];         // 4x 4 KB

  const int lane = threadIdx.x & 31;
  const int wave = threadIdx.x >> 5;
  const int pix0 = blockIdx.x * PIX_PER_BLK + wave * PIX_PER_WAVE;
  const int pix  = pix0 + (lane & 15);          // this lane's GEMM row (A operand)

  __bf16* qS = sQ[wave];
  __bf16* kS = sK[wave];
  __bf16* vS = sV[wave];
  __bf16* oS = sO[wave];

  // ---- Q projection --------------------------------------------------------
  tdm_load_32k(wbf + 0 * 16384, sW);
  gemm_proj(Q + (size_t)pix * DMODEL, sW, in_b, qS, 0, lane);

  // ---- K projection (3 tokens -> rows t*16+p) ------------------------------
  tdm_load_32k(wbf + 1 * 16384, sW);
#pragma unroll
  for (int mt = 0; mt < LKTOK; ++mt)
    gemm_proj(K + ((size_t)pix * LKTOK + mt) * DMODEL, sW, in_b + 128, kS, mt * 16, lane);

  // ---- V projection --------------------------------------------------------
  tdm_load_32k(wbf + 2 * 16384, sW);
#pragma unroll
  for (int mt = 0; mt < LKTOK; ++mt)
    gemm_proj(V + ((size_t)pix * LKTOK + mt) * DMODEL, sW, in_b + 256, vS, mt * 16, lane);

  __syncthreads();   // q/k/v staging written cross-lane -> make visible

  // ---- attention: 1x3 softmax per (pixel, head); 4 tasks per lane ----------
  {
    const int p     = lane & 15;
    const int hbase = (lane >> 4) * 4;
#pragma unroll
    for (int hh = 0; hh < 4; ++hh) {
      const int h = hbase + hh;
      float qv[16];
      load16(qS + p * 128 + h * 16, qv);
      float sc[3];
#pragma unroll
      for (int t = 0; t < 3; ++t) {
        float kv[16];
        load16(kS + (t * 16 + p) * 128 + h * 16, kv);
        float acc = 0.f;
#pragma unroll
        for (int d = 0; d < 16; ++d) acc += qv[d] * kv[d];
        sc[t] = acc * 0.25f;          // 1/sqrt(HD)
      }
      const float m  = fmaxf(sc[0], fmaxf(sc[1], sc[2]));
      const float e0 = __expf(sc[0] - m);
      const float e1 = __expf(sc[1] - m);
      const float e2 = __expf(sc[2] - m);
      const float inv = 1.f / (e0 + e1 + e2);
      const float w0 = e0 * inv, w1 = e1 * inv, w2 = e2 * inv;

      float v0[16], v1[16], v2[16];
      load16(vS + (0 * 16 + p) * 128 + h * 16, v0);
      load16(vS + (1 * 16 + p) * 128 + h * 16, v1);
      load16(vS + (2 * 16 + p) * 128 + h * 16, v2);
#pragma unroll
      for (int d = 0; d < 16; ++d)
        oS[p * 128 + h * 16 + d] = (__bf16)(w0 * v0[d] + w1 * v1[d] + w2 * v2[d]);
    }
  }

  // ---- output projection: o(16x128) @ Wo^T + out_b -> global ---------------
  tdm_load_32k(wbf + 3 * 16384, sW);   // its barriers also publish sO
  {
    v16bf a[4];
    const __bf16* orow = oS + (lane & 15) * 128;
#pragma unroll
    for (int kb = 0; kb < 4; ++kb) a[kb] = make_a_from_bf16(orow, kb, lane);
    for (int j = 0; j < 8; ++j) {
      const float bj = out_b[j * 16 + (lane & 15)];
      v8f c = {bj, bj, bj, bj, bj, bj, bj, bj};
#pragma unroll
      for (int kb = 0; kb < 4; ++kb) {
        v16bf b = make_b_from_lds(sW, j, kb, lane);
        c = __builtin_amdgcn_wmma_f32_16x16x32_bf16(false, a[kb], false, b,
                                                    (short)0, c, false, false);
      }
      const int col = j * 16 + (lane & 15);
      const int r0  = (lane >> 4) * 8;
#pragma unroll
      for (int v = 0; v < 8; ++v)
        out[(size_t)(pix0 + r0 + v) * DMODEL + col] = c[v];
    }
  }
}

// ---------------- host launch ------------------------------------------------
extern "C" void kernel_launch(void* const* d_in, const int* in_sizes, int n_in,
                              void* d_out, int out_size, void* d_ws, size_t ws_size,
                              hipStream_t stream) {
  const float* Q     = (const float*)d_in[0];
  const float* K     = (const float*)d_in[1];
  const float* V     = (const float*)d_in[2];
  const float* in_w  = (const float*)d_in[3];
  const float* in_b  = (const float*)d_in[4];
  const float* out_w = (const float*)d_in[5];
  const float* out_b = (const float*)d_in[6];
  float*       out   = (float*)d_out;
  __bf16*      ws    = (__bf16*)d_ws;            // 128 KB bf16 weights

  // 32768 packed bf16 pairs total
  cvt_weights_kernel<<<128, 256, 0, stream>>>(in_w, out_w, ws);
  fused_mha_kernel<<<NPIX / PIX_PER_BLK, 128, 0, stream>>>(
      Q, K, V, in_b, out_b, ws, out);
}
